// SubModel_10857677324964
// MI455X (gfx1250) — matrix operand order repaired and checked by reading, not compiled, not run on hardware
//
#include <hip/hip_runtime.h>
#include <hip/hip_bf16.h>
#include <math.h>

// ---------------------------------------------------------------------------
// Sizes (from reference): B=64, S=512, D=128, DFF=256, HID=230 (pad to 240).
// Fused per-64-token-tile transformer block using v_wmma_f32_16x16x32_f16.
// ---------------------------------------------------------------------------

typedef _Float16 half_t;
typedef __attribute__((ext_vector_type(16))) _Float16 v16h;
typedef __attribute__((ext_vector_type(8)))  float    v8f;

constexpr int TD   = 128;   // model dim
constexpr int TDFF = 256;   // ffn dim
constexpr int THID = 230;   // final out dim
constexpr int THIDP= 240;   // padded to 16
constexpr int TT   = 64;    // tokens per block tile
constexpr int LDF  = 132;   // f32 LDS row stride (pad vs 64-bank conflicts)
constexpr int LDH  = 264;   // f16 LDS row stride (max K=256 + pad)

// ---------------------------------------------------------------------------
// Weight packing: W is [N][K] row-major (out = x @ W^T). We store the WMMA
// B-matrix (K x N) fragments pre-swizzled so a fragment is 16 contiguous f16
// per lane:  dst[((ct*(K/32)+k0)*32 + lane)*16 + i].
// Lane layout: col = ct*16 + lane%16, half = lane/16,
//   element i (pair p=i/2, j=i%2) -> kk = k0*32 + (p<4?0:16) + half*8 + 2*(p&3) + j.
// ---------------------------------------------------------------------------
__global__ void pack_weight(const float* __restrict__ W, half_t* __restrict__ dst,
                            int N, int K, int realN) {
  int idx = blockIdx.x * 256 + threadIdx.x;
  if (idx >= N * K) return;
  int i    = idx & 15;
  int lane = (idx >> 4) & 31;
  int fk   = idx >> 9;              // ct*(K/32) + k0
  int ks   = K >> 5;
  int ct   = fk / ks;
  int k0   = fk - ct * ks;
  int col  = ct * 16 + (lane & 15);
  int hf   = lane >> 4;
  int p    = i >> 1, j = i & 1;
  int kk   = k0 * 32 + ((p < 4) ? 0 : 16) + hf * 8 + 2 * (p & 3) + j;
  float v  = (col < realN) ? W[col * K + kk] : 0.0f;
  dst[idx] = (half_t)v;
}

// ---------------------------------------------------------------------------
// Per-batch attention collapse. softmax(mask outer-product) has only two
// distinct rows per batch; fold the (linear) v- and fc-projections through:
//   rowU_fc = ((sumInp/S) @ wv^T + bv) @ fc^T + fcb
//   rowW_fc = (((e-1)*sumInp1 + sumInp)/Z @ wv^T + bv) @ fc^T + fcb,
//   Z = cnt*(e-1) + S.
// ---------------------------------------------------------------------------
__global__ __launch_bounds__(256)
void batch_stats(const float* __restrict__ inp, const int* __restrict__ mask,
                 const float* __restrict__ wv, const float* __restrict__ bv,
                 const float* __restrict__ fcw, const float* __restrict__ fcb,
                 float* __restrict__ rowU, float* __restrict__ rowW) {
  __shared__ float s_p[2][TD], s_p1[2][TD];
  __shared__ float s_yU[TD], s_yW[TD], s_rU[TD], s_rW[TD];
  __shared__ int s_cnt;
  const int b = blockIdx.x, tid = threadIdx.x;
  const int d = tid & 127, part = tid >> 7;
  if (tid == 0) s_cnt = 0;
  __syncthreads();

  float a0 = 0.f, a1 = 0.f; int c = 0;
  const float* base = inp + (size_t)b * 512 * TD;
  const int*   mrow = mask + b * 512;
  for (int t = part * 256; t < part * 256 + 256; ++t) {
    float v = base[t * TD + d];
    int   m = mrow[t];
    a0 += v;
    if (m) a1 += v;
    if (d == 0) c += m;
  }
  s_p[part][d] = a0; s_p1[part][d] = a1;
  if (d == 0) atomicAdd(&s_cnt, c);
  __syncthreads();

  if (tid < TD) {
    float s  = s_p[0][tid] + s_p[1][tid];
    float s1 = s_p1[0][tid] + s_p1[1][tid];
    const float em1 = 1.7182818284590452f;      // e - 1
    float Z = (float)s_cnt * em1 + 512.0f;
    s_yU[tid] = s * (1.0f / 512.0f);
    s_yW[tid] = (em1 * s1 + s) / Z;
  }
  __syncthreads();
  if (tid < TD) {                                // through wv^T (+bv)
    float accU = bv[tid], accW = bv[tid];
    const float* w = wv + tid * TD;
    for (int k = 0; k < TD; ++k) { accU += s_yU[k] * w[k]; accW += s_yW[k] * w[k]; }
    s_rU[tid] = accU; s_rW[tid] = accW;
  }
  __syncthreads();
  if (tid < TD) {                                // through fc_w^T (+fcb)
    float accU = fcb[tid], accW = fcb[tid];
    const float* w = fcw + tid * TD;
    for (int k = 0; k < TD; ++k) { accU += s_rU[k] * w[k]; accW += s_rW[k] * w[k]; }
    rowU[b * TD + tid] = accU;
    rowW[b * TD + tid] = accW;
  }
}

// ---------------------------------------------------------------------------
// WMMA helpers — K statically unrolled (K/32 WMMA steps).
// ---------------------------------------------------------------------------
template<int K>
__device__ inline void gemm_acc(const half_t* __restrict__ act,       // LDS, stride LDH
                                const half_t* __restrict__ wpack,     // global packed
                                int rt, int ct, int lane, v8f& acc) {
  constexpr int ks = K >> 5;
  const int row = rt * 16 + (lane & 15);
  const int hf  = lane >> 4;
  const half_t* arow = act + row * LDH;
#pragma unroll
  for (int k0 = 0; k0 < ks; ++k0) {
    // A fragment: 16-bit A 16x32 layout — lane(row) gathers K pairs from LDS.
    union { unsigned u[8]; v16h v; } a;
    const half_t* ab = arow + k0 * 32;
#pragma unroll
    for (int p = 0; p < 8; ++p) {
      int kk = ((p < 4) ? 0 : 16) + hf * 8 + 2 * (p & 3);
      a.u[p] = *(const unsigned*)(ab + kk);       // ds_load_b32 (two f16)
    }
    // B fragment: pre-swizzled, contiguous 32B per lane.
    v16h bf = *(const v16h*)(wpack + ((size_t)((ct * ks + k0) * 32 + lane) << 4));
    acc = __builtin_amdgcn_wmma_f32_16x16x32_f16(false, a.v, false, bf,
                                                 (short)0, acc, false, false);
  }
}

// GEMM stage -> relu -> f16 LDS.  N = 16*CT_TILES outputs, fully unrolled.
template<int K, int CT_TILES>
__device__ inline void gemm_relu_f16(const half_t* __restrict__ actIn,
                                     const half_t* __restrict__ wpack,
                                     const float* __restrict__ bias,
                                     half_t* __restrict__ dstH,
                                     int wave, int lane) {
  constexpr int ITEMS = 4 * CT_TILES;
  constexpr int REPS  = ITEMS / 8;
  const int hf = lane >> 4;
#pragma unroll
  for (int rep = 0; rep < REPS; ++rep) {
    const int it = wave + rep * 8;
    const int rt = it & 3, ct = it >> 2;
    v8f acc = {};
    gemm_acc<K>(actIn, wpack, rt, ct, lane, acc);
    const int col = ct * 16 + (lane & 15);
    const float bb = bias[col];
#pragma unroll
    for (int j = 0; j < 8; ++j)
      dstH[(rt * 16 + j + 8 * hf) * LDH + col] = (half_t)fmaxf(acc[j] + bb, 0.f);
  }
}

// GEMM stage -> f32 LDS (bias scaled by bscale).
template<int K, int CT_TILES>
__device__ inline void gemm_f32(const half_t* __restrict__ actIn,
                                const half_t* __restrict__ wpack,
                                const float* __restrict__ bias, float bscale,
                                float* __restrict__ dstF,
                                int wave, int lane) {
  constexpr int ITEMS = 4 * CT_TILES;
  constexpr int REPS  = ITEMS / 8;
  const int hf = lane >> 4;
#pragma unroll
  for (int rep = 0; rep < REPS; ++rep) {
    const int it = wave + rep * 8;
    const int rt = it & 3, ct = it >> 2;
    v8f acc = {};
    gemm_acc<K>(actIn, wpack, rt, ct, lane, acc);
    const int col = ct * 16 + (lane & 15);
    const float bb = bscale * bias[col];
#pragma unroll
    for (int j = 0; j < 8; ++j)
      dstF[(rt * 16 + j + 8 * hf) * LDF + col] = acc[j] + bb;
  }
}

// L2 prefetch of a packed-weight region (global_prefetch_b8), 256B lines.
__device__ inline void prefetch_region(const void* p, int bytes, int tid) {
  const char* c = (const char*)p;
  for (int off = tid * 256; off < bytes; off += 256 * 256)
    __builtin_prefetch(c + off, 0, 1);
}

// ---------------------------------------------------------------------------
// LayerNorm variants (thread-per-token; LN cost is negligible vs GEMMs)
// ---------------------------------------------------------------------------
__device__ inline void ln_torch_f(const float* src, float* dstF, half_t* dstH,
                                  const float* __restrict__ g,
                                  const float* __restrict__ bt, int tid) {
  if (tid < TT) {
    const float* x = src + tid * LDF;
    float s = 0.f;
    for (int d = 0; d < TD; ++d) s += x[d];
    float mean = s * (1.0f / TD);
    float vs = 0.f;
    for (int d = 0; d < TD; ++d) { float t = x[d] - mean; vs += t * t; }
    float inv = rsqrtf(vs * (1.0f / TD) + 1e-5f);
    float* df = dstF + tid * LDF;
    half_t* dh = dstH + tid * LDH;
    for (int d = 0; d < TD; ++d) {
      float v = g[d] * (x[d] - mean) * inv + bt[d];
      df[d] = v; dh[d] = (half_t)v;
    }
  }
}

__device__ inline void ln_custom_f(const float* src, float* dstF, half_t* dstH,
                                   const float* __restrict__ a,
                                   const float* __restrict__ bt, int tid) {
  if (tid < TT) {
    const float* x = src + tid * LDF;
    float s = 0.f;
    for (int d = 0; d < TD; ++d) s += x[d];
    float mean = s * (1.0f / TD);
    float vs = 0.f;
    for (int d = 0; d < TD; ++d) { float t = x[d] - mean; vs += t * t; }
    float r = 1.0f / (sqrtf(vs * (1.0f / (TD - 1))) + 1e-6f);
    float* df = dstF + tid * LDF;
    half_t* dh = dstH + tid * LDH;
    for (int d = 0; d < TD; ++d) {
      float v = a[d] * (x[d] - mean) * r + bt[d];
      df[d] = v; dh[d] = (half_t)v;
    }
  }
}

// ---------------------------------------------------------------------------
// Fused transformer block: one 64-token tile per workgroup, 8 waves.
// LDS: outF(33 KB f32) | actF(33 KB f32) | actH(33 KB f16) | actH2(33 KB f16)
// ---------------------------------------------------------------------------
__global__ __launch_bounds__(256)
void fused_block(const float* __restrict__ inp, const int* __restrict__ mask,
                 const float* __restrict__ rowU, const float* __restrict__ rowW,
                 const half_t* __restrict__ ws_w1,  const half_t* __restrict__ ws_w2,
                 const half_t* __restrict__ ws_proj,const half_t* __restrict__ ws_ffn,
                 const half_t* __restrict__ ws_pint,const half_t* __restrict__ ws_final,
                 const float* __restrict__ b1,    const float* __restrict__ b2,
                 const float* __restrict__ projb, const float* __restrict__ ffnb,
                 const float* __restrict__ pintb, const float* __restrict__ finalb,
                 const float* __restrict__ g_mha, const float* __restrict__ bb_mha,
                 const float* __restrict__ a_int, const float* __restrict__ b_int,
                 const float* __restrict__ a_ctx, const float* __restrict__ b_ctx,
                 const float* __restrict__ a_fin, const float* __restrict__ b_fin,
                 float* __restrict__ out) {
  extern __shared__ char smem[];
  float*  outF  = (float*)smem;                 // [TT][LDF] GEMM f32 outputs / pre-LN
  float*  actF  = outF + TT * LDF;              // [TT][LDF] current f32 activation
  half_t* actH  = (half_t*)(actF + TT * LDF);   // [TT][LDH] f16 GEMM A-input
  half_t* actH2 = actH + TT * LDH;              // [TT][LDH] f16 DFF activations

  const int tid  = threadIdx.x;
  const int lane = tid & 31;
  const int wave = tid >> 5;
  const int hf   = lane >> 4;
  const int blk  = blockIdx.x;
  const int b    = blk >> 3;                    // 8 tiles per batch
  const int tok0 = blk * TT;                    // flat token base

  // warm L2 for the first two weight stages while we do stage 1 + LN
  prefetch_region(ws_w1, TDFF * TD * 2, tid);
  prefetch_region(ws_w2, TD * TDFF * 2, tid);

  // --- stage 1: attention output (mask-select) + residual -------------------
  for (int e = tid; e < TT * TD; e += 256) {
    int t = e >> 7, d = e & 127;
    int gt = tok0 + t;
    float h1 = mask[gt] ? rowW[b * TD + d] : rowU[b * TD + d];
    outF[t * LDF + d] = h1 + inp[(size_t)gt * TD + d];
  }
  __syncthreads();
  ln_torch_f(outF, actF, actH, g_mha, bb_mha, tid);          // x1
  __syncthreads();

  // --- GEMM1: h2 = relu(x1 @ w1^T + b1)   K=128 N=256 -> actH2 (f16) --------
  gemm_relu_f16<TD, TDFF / 16>(actH, ws_w1, b1, actH2, wave, lane);
  prefetch_region(ws_proj, TD * TD * 2, tid);
  prefetch_region(ws_ffn, TDFF * TD * 2, tid);
  __syncthreads();

  // --- GEMM2: h3 = h2 @ w2^T + b2         K=256 N=128 -> outF (f32) ---------
  gemm_f32<TDFF, TD / 16>(actH2, ws_w2, b2, 1.0f, outF, wave, lane);
  __syncthreads();
  for (int e = tid; e < TT * TD; e += 256) {                 // + x1
    int t = e >> 7, d = e & 127;
    outF[t * LDF + d] += actF[t * LDF + d];
  }
  __syncthreads();
  ln_custom_f(outF, actF, actH, a_int, b_int, tid);          // x2
  __syncthreads();

  // --- shared projection: (x0 + x2) @ proj^T + 2*proj_b ---------------------
  for (int e = tid; e < TT * TD; e += 256) {
    int t = e >> 7, d = e & 127;
    int gt = tok0 + t;
    actH[t * LDH + d] = (half_t)(inp[(size_t)gt * TD + d] + actF[t * LDF + d]);
  }
  __syncthreads();
  gemm_f32<TD, TD / 16>(actH, ws_proj, projb, 2.0f, outF, wave, lane);  // GEMM3
  prefetch_region(ws_pint, TD * TDFF * 2, tid);
  prefetch_region(ws_final, THIDP * TD * 2, tid);
  __syncthreads();
  ln_custom_f(outF, actF, actH, a_ctx, b_ctx, tid);          // x3 (res_inp)
  __syncthreads();

  // --- GEMM4: h5 = relu(x3 @ ffn^T + ffnb)  K=128 N=256 -> actH2 ------------
  gemm_relu_f16<TD, TDFF / 16>(actH, ws_ffn, ffnb, actH2, wave, lane);
  __syncthreads();

  // --- GEMM5: h6 = h5 @ proj_int^T + b      K=256 N=128 -> outF -------------
  gemm_f32<TDFF, TD / 16>(actH2, ws_pint, pintb, 1.0f, outF, wave, lane);
  __syncthreads();
  for (int e = tid; e < TT * TD; e += 256) {                 // + x3
    int t = e >> 7, d = e & 127;
    outF[t * LDF + d] += actF[t * LDF + d];
  }
  __syncthreads();
  ln_custom_f(outF, actF, actH, a_fin, b_fin, tid);          // x4
  __syncthreads();

  // --- GEMM6: out = x4 @ final^T + final_b  K=128 N=240(->230) to global ----
  // 60 tiles: wave-uniform guard (EXEC untouched around WMMA).
#pragma unroll
  for (int rep = 0; rep < 8; ++rep) {
    const int it = wave + rep * 8;
    if (it < 4 * (THIDP / 16)) {
      const int rt = it & 3, ct = it >> 2;
      v8f acc = {};
      gemm_acc<TD>(actH, ws_final, rt, ct, lane, acc);
      const int col = ct * 16 + (lane & 15);
      if (col < THID) {
        const float bias = finalb[col];
#pragma unroll
        for (int j = 0; j < 8; ++j)
          out[(size_t)(tok0 + rt * 16 + j + 8 * hf) * THID + col] = acc[j] + bias;
      }
    }
  }
}

// ---------------------------------------------------------------------------
// Host launcher. d_in order: inp, mask, wq,bq, wk,bk, wv,bv, fc_w,fc_b,
// ln_mha_g,ln_mha_b, w1,b1, w2,b2, ln_int_a,ln_int_b, proj_w,proj_b,
// ln_ctx_a,ln_ctx_b, ffn_w,ffn_b, proj_int_w,proj_int_b, ln_fin_a,ln_fin_b,
// final_w,final_b.  (wq/bq/wk/bk are dead code in the reference.)
// ---------------------------------------------------------------------------
extern "C" void kernel_launch(void* const* d_in, const int* in_sizes, int n_in,
                              void* d_out, int out_size, void* d_ws, size_t ws_size,
                              hipStream_t stream) {
  (void)in_sizes; (void)n_in; (void)out_size; (void)ws_size;
  const float* inp    = (const float*)d_in[0];
  const int*   mask   = (const int*)  d_in[1];
  const float* wv     = (const float*)d_in[6];
  const float* bv     = (const float*)d_in[7];
  const float* fcw    = (const float*)d_in[8];
  const float* fcb    = (const float*)d_in[9];
  const float* g_mha  = (const float*)d_in[10];
  const float* bb_mha = (const float*)d_in[11];
  const float* w1     = (const float*)d_in[12];
  const float* b1     = (const float*)d_in[13];
  const float* w2     = (const float*)d_in[14];
  const float* b2     = (const float*)d_in[15];
  const float* a_int  = (const float*)d_in[16];
  const float* b_int  = (const float*)d_in[17];
  const float* projw  = (const float*)d_in[18];
  const float* projb  = (const float*)d_in[19];
  const float* a_ctx  = (const float*)d_in[20];
  const float* b_ctx  = (const float*)d_in[21];
  const float* ffnw   = (const float*)d_in[22];
  const float* ffnb   = (const float*)d_in[23];
  const float* pintw  = (const float*)d_in[24];
  const float* pintb  = (const float*)d_in[25];
  const float* a_fin  = (const float*)d_in[26];
  const float* b_fin  = (const float*)d_in[27];
  const float* finalw = (const float*)d_in[28];
  const float* finalb = (const float*)d_in[29];
  float* out = (float*)d_out;

  // workspace layout (f16 packed weights, then f32 attention rows)
  half_t* ws_w1    = (half_t*)d_ws;                 // 256*128
  half_t* ws_w2    = ws_w1   + TDFF * TD;           // 128*256
  half_t* ws_proj  = ws_w2   + TD * TDFF;           // 128*128
  half_t* ws_ffn   = ws_proj + TD * TD;             // 256*128
  half_t* ws_pint  = ws_ffn  + TDFF * TD;           // 128*256
  half_t* ws_final = ws_pint + TD * TDFF;           // 240*128
  float*  rowU     = (float*)(ws_final + THIDP * TD);
  float*  rowW     = rowU + 64 * TD;

  pack_weight<<<(TDFF * TD) / 256, 256, 0, stream>>>(w1,    ws_w1,    TDFF, TD,   TDFF);
  pack_weight<<<(TD * TDFF) / 256, 256, 0, stream>>>(w2,    ws_w2,    TD,   TDFF, TD);
  pack_weight<<<(TD * TD)   / 256, 256, 0, stream>>>(projw, ws_proj,  TD,   TD,   TD);
  pack_weight<<<(TDFF * TD) / 256, 256, 0, stream>>>(ffnw,  ws_ffn,   TDFF, TD,   TDFF);
  pack_weight<<<(TD * TDFF) / 256, 256, 0, stream>>>(pintw, ws_pint,  TD,   TDFF, TD);
  pack_weight<<<(THIDP * TD)/ 256, 256, 0, stream>>>(finalw,ws_final, THIDP,TD,   THID);

  batch_stats<<<64, 256, 0, stream>>>(inp, mask, wv, bv, fcw, fcb, rowU, rowW);

  const size_t shbytes = (size_t)TT * LDF * 4 * 2 + (size_t)TT * LDH * 2 * 2; // 132 KB (WGP has 320 KB)
  fused_block<<<512, 256, shbytes, stream>>>(inp, mask, rowU, rowW,
      ws_w1, ws_w2, ws_proj, ws_ffn, ws_pint, ws_final,
      b1, b2, projb, ffnb, pintb, finalb,
      g_mha, bb_mha, a_int, b_int, a_ctx, b_ctx, a_fin, b_fin, out);
}
